// ParentPredictionGNN_75093208203878
// MI455X (gfx1250) — matrix operand order repaired and checked by reading, not compiled, not run on hardware
//
#include <hip/hip_runtime.h>
#include <hip/hip_bf16.h>

// ParentPredictionGNN for MI455X (gfx1250, wave32, WMMA).
// GEMMs: f16 WMMA 16x16x32, f32 accum. Each wave computes a 32x32 register tile
// (2x2 WMMA tiles) so each A/B fragment is reused twice: 8 b128 loads per
// 4 WMMAs instead of 16. B operands kept as B^T row-major so fragments are
// contiguous 16B loads per lane (ISA 7.12.2 layout).
// Score masking (gender/generation/bilinear-bias) is fused into the score-GEMM
// epilogue, eliminating a full 268MB read+write pass per head.

#define NNODE 16384
#define EEDGE 262144
#define QQ    4096
#define FEAT  64
#define HIDD  256
#define EMBD  128

typedef __attribute__((ext_vector_type(16))) _Float16 v16h;
typedef __attribute__((ext_vector_type(8)))  float    v8f;

union FragH { v16h v; uint4 u[2]; };

// ---------------------------------------------------------------- GEMM (WMMA)
// C[M,Nn] = A[M,K] @ B[K,Nn], given BT[Nn,K] row-major (both f16).
// Block: 256 threads = 8 waves arranged 2(M) x 4(N); block tile = 64M x 128N.
// Wave tile = 32M x 32N (2x2 accumulators). Requires M%64==0, Nn%128==0, K%32==0.
template<int RELU, int ACC, int MASK>
__global__ __launch_bounds__(256) void k_gemm(
    const _Float16* __restrict__ A, const _Float16* __restrict__ BT,
    const float* __restrict__ bias, float* __restrict__ C32,
    _Float16* __restrict__ C16, int M, int Nn, int K,
    const int* __restrict__ gender, const int* __restrict__ gen,
    const int* __restrict__ qidx, const float* __restrict__ bb, int TG)
{
    const int lane = threadIdx.x & 31;
    const int wv   = threadIdx.x >> 5;
    const int wm   = wv >> 2;               // 0..1
    const int wn   = wv & 3;                // 0..3
    const int tm0  = blockIdx.y * 4 + wm * 2;   // first M tile (of 16 rows)
    const int tn0  = blockIdx.x * 8 + wn * 2;   // first N tile (of 16 cols)
    const int hl   = lane >> 4;             // lane half (0/1)
    const int l15  = lane & 15;

    const _Float16* aRow0 = A  + (size_t)(tm0 * 16 + l15) * K;
    const _Float16* aRow1 = aRow0 + (size_t)16 * K;
    const _Float16* bRow0 = BT + (size_t)(tn0 * 16 + l15) * K;
    const _Float16* bRow1 = bRow0 + (size_t)16 * K;

    v8f acc[2][2] = {};
    for (int kb = 0; kb < K; kb += 32) {
        FragH fa[2], fb[2];
        // A frag: elems 0..7 -> k = kb+8*hl..+7 ; elems 8..15 -> kb+16+8*hl..+7
        fa[0].u[0] = *(const uint4*)(aRow0 + kb + 8 * hl);
        fa[0].u[1] = *(const uint4*)(aRow0 + kb + 16 + 8 * hl);
        fa[1].u[0] = *(const uint4*)(aRow1 + kb + 8 * hl);
        fa[1].u[1] = *(const uint4*)(aRow1 + kb + 16 + 8 * hl);
        // B frag: elems 0..7 -> k = kb+8*hl..+7 ; elems 8..15 -> kb+16+8*hl..+7
        fb[0].u[0] = *(const uint4*)(bRow0 + kb + 8 * hl);
        fb[0].u[1] = *(const uint4*)(bRow0 + kb + 16 + 8 * hl);
        fb[1].u[0] = *(const uint4*)(bRow1 + kb + 8 * hl);
        fb[1].u[1] = *(const uint4*)(bRow1 + kb + 16 + 8 * hl);
#pragma unroll
        for (int i = 0; i < 2; ++i)
#pragma unroll
            for (int j = 0; j < 2; ++j)
                acc[i][j] = __builtin_amdgcn_wmma_f32_16x16x32_f16(
                    false, fa[i].v, false, fb[j].v, (short)0, acc[i][j], false, false);
    }

    const float bbv = MASK ? bb[0] : 0.f;
#pragma unroll
    for (int i = 0; i < 2; ++i) {
        int qg[8];
        if (MASK) {
#pragma unroll
            for (int r = 0; r < 8; ++r)
                qg[r] = gen[qidx[(tm0 + i) * 16 + r + 8 * hl]];
        }
#pragma unroll
        for (int j = 0; j < 2; ++j) {
            const int n = (tn0 + j) * 16 + l15;
            const float bv = (!MASK && bias) ? bias[n] : 0.f;
            int gd = 0, gn = 0;
            if (MASK) { gd = gender[n]; gn = gen[n]; }
#pragma unroll
            for (int r = 0; r < 8; ++r) {
                const int m = (tm0 + i) * 16 + r + 8 * hl;
                const size_t off = (size_t)m * Nn + n;
                float v = acc[i][j][r] + bv;
                if (ACC)  v += C32[off];
                if (RELU) v = v > 0.f ? v : 0.f;
                if (MASK) {
                    const int diff = gn - qg[r];         // integer generations
                    const bool valid = (gd == TG) && diff >= 1 && diff <= 2;
                    v = valid ? v + bbv : -__builtin_inff();
                }
                if (C32) C32[off] = v;
                if (C16) C16[off] = (_Float16)v;
            }
        }
    }
}

// ------------------------------------------------------------- small helpers
__global__ void k_f2h(const float* __restrict__ in, _Float16* __restrict__ out, long n) {
    long i = (long)blockIdx.x * blockDim.x + threadIdx.x;
    if (i < n) out[i] = (_Float16)in[i];
}

// out[c*R + r] = in[r*C + c]   (f32 [R,C] -> f16 [C,R])
__global__ void k_f2h_t(const float* __restrict__ in, _Float16* __restrict__ out, int R, int C) {
    int i = blockIdx.x * blockDim.x + threadIdx.x;
    if (i < R * C) {
        int r = i / C, c = i % C;
        out[(size_t)c * R + r] = (_Float16)in[i];
    }
}

__global__ void k_fill0(unsigned* __restrict__ p, long n) {
    long i = (long)blockIdx.x * blockDim.x + threadIdx.x;
    if (i < n) p[i] = 0u;
}

__global__ void k_gather16(const _Float16* __restrict__ emb, const int* __restrict__ idx,
                           _Float16* __restrict__ out, int Q, int d) {
    int i = blockIdx.x * blockDim.x + threadIdx.x;
    if (i < Q * d) {
        int r = i / d, c = i % d;
        out[i] = emb[(size_t)idx[r] * d + c];
    }
}

// ------------------------------------------------------------- attention ops
__device__ __forceinline__ unsigned f2o(float f) {
    unsigned u = __float_as_uint(f);
    return (u & 0x80000000u) ? ~u : (u | 0x80000000u);
}
__device__ __forceinline__ float o2f(unsigned u) {
    return __uint_as_float((u & 0x80000000u) ? (u & 0x7FFFFFFFu) : ~u);
}
__device__ __forceinline__ float lrelu(float v) { return v > 0.f ? v : 0.2f * v; }

// a_src[n,h] = sum_c z[n,h,c]*att_src[h,c] ; same for dst
__global__ void k_attn(const float* __restrict__ z, const float* __restrict__ att_s,
                       const float* __restrict__ att_d, float* __restrict__ a_src,
                       float* __restrict__ a_dst, int N, int H, int C) {
    int i = blockIdx.x * blockDim.x + threadIdx.x;
    if (i >= N * H) return;
    int n = i / H, h = i % H;
    const float* zp = z + (size_t)n * H * C + (size_t)h * C;
    float s0 = 0.f, s1 = 0.f;
    for (int c = 0; c < C; ++c) {
        float zv = zp[c];
        s0 += zv * att_s[h * C + c];
        s1 += zv * att_d[h * C + c];
    }
    a_src[i] = s0; a_dst[i] = s1;
}

__device__ __forceinline__ void edge_nodes(const int* ei, int e, int E, int& s, int& d) {
    if (e < E) { s = ei[e]; d = ei[E + e]; } else { s = e - E; d = e - E; }
}

__global__ void k_edge_max(const int* __restrict__ ei, const float* __restrict__ a_src,
                           const float* __restrict__ a_dst, unsigned* __restrict__ m_enc,
                           int E, int TOT, int H) {
    long li = (long)blockIdx.x * blockDim.x + threadIdx.x;
    if (li >= (long)TOT * H) return;
    int i = (int)li, e = i / H, h = i % H, s, d;
    edge_nodes(ei, e, E, s, d);
    float v = lrelu(a_src[s * H + h] + a_dst[d * H + h]);
    atomicMax(&m_enc[d * H + h], f2o(v));
}

__global__ void k_edge_sum(const int* __restrict__ ei, const float* __restrict__ a_src,
                           const float* __restrict__ a_dst, const unsigned* __restrict__ m_enc,
                           float* __restrict__ ssum, int E, int TOT, int H) {
    long li = (long)blockIdx.x * blockDim.x + threadIdx.x;
    if (li >= (long)TOT * H) return;
    int i = (int)li, e = i / H, h = i % H, s, d;
    edge_nodes(ei, e, E, s, d);
    float v = lrelu(a_src[s * H + h] + a_dst[d * H + h]);
    atomicAdd(&ssum[d * H + h], __expf(v - o2f(m_enc[d * H + h])));
}

__global__ void k_edge_acc(const int* __restrict__ ei, const float* __restrict__ a_src,
                           const float* __restrict__ a_dst, const unsigned* __restrict__ m_enc,
                           const float* __restrict__ ssum, const float* __restrict__ z,
                           float* __restrict__ outp, int E, int TOT, int H, int C) {
    long li = (long)blockIdx.x * blockDim.x + threadIdx.x;
    if (li >= (long)TOT * H) return;
    int i = (int)li, e = i / H, h = i % H, s, d;
    edge_nodes(ei, e, E, s, d);
    float v = lrelu(a_src[s * H + h] + a_dst[d * H + h]);
    float alpha = __expf(v - o2f(m_enc[d * H + h])) / ssum[d * H + h];
    const float* zp = z + (size_t)s * H * C + (size_t)h * C;
    float* op = outp + (size_t)d * H * C + (size_t)h * C;
    for (int c = 0; c < C; ++c) atomicAdd(&op[c], zp[c] * alpha);
}

// -------------------------------------------------------------- layernorm
// one wave per row (wave32); out = LN(in + inbias)*g + b (+relu), f32/f16 sinks
__global__ __launch_bounds__(256) void k_ln(const float* __restrict__ in,
    const float* __restrict__ inbias, const float* __restrict__ g,
    const float* __restrict__ b, float* __restrict__ out32,
    _Float16* __restrict__ out16, int rows, int d, int relu)
{
    int wv = threadIdx.x >> 5, lane = threadIdx.x & 31;
    int row = blockIdx.x * 8 + wv;
    if (row >= rows) return;
    const float* rp = in + (size_t)row * d;
    const int per = d / 32;           // 8 (d=256) or 4 (d=128)
    float vals[8];
    float s = 0.f;
    for (int j = 0; j < per; ++j) {
        int col = lane * per + j;
        float v = rp[col] + (inbias ? inbias[col] : 0.f);
        vals[j] = v; s += v;
    }
    for (int o = 16; o > 0; o >>= 1) s += __shfl_xor(s, o, 32);
    const float mu = s / (float)d;
    float vs = 0.f;
    for (int j = 0; j < per; ++j) { float c = vals[j] - mu; vs += c * c; }
    for (int o = 16; o > 0; o >>= 1) vs += __shfl_xor(vs, o, 32);
    const float inv = rsqrtf(vs / (float)d + 1e-5f);
    for (int j = 0; j < per; ++j) {
        int col = lane * per + j;
        float v = (vals[j] - mu) * inv * g[col] + b[col];
        if (relu) v = v > 0.f ? v : 0.f;
        size_t off = (size_t)row * d + col;
        if (out32) out32[off] = v;
        if (out16) out16[off] = (_Float16)v;
    }
}

// ================================================================== host
extern "C" void kernel_launch(void* const* d_in, const int* in_sizes, int n_in,
                              void* d_out, int out_size, void* d_ws, size_t ws_size,
                              hipStream_t stream) {
    (void)in_sizes; (void)n_in; (void)out_size; (void)ws_size;

    // inputs per setup_inputs() insertion order (params flattened depth-first)
    const float* x      = (const float*)d_in[0];
    const int*   ei     = (const int*)d_in[1];
    const int*   qidx   = (const int*)d_in[2];
    const int*   gender = (const int*)d_in[3];
    const int*   gener  = (const int*)d_in[4];
    const float* wip    = (const float*)d_in[5];
    const float* bip    = (const float*)d_in[6];
    const float* gatW[4] = {(const float*)d_in[7],  (const float*)d_in[11],
                            (const float*)d_in[15], (const float*)d_in[19]};
    const float* atS[4]  = {(const float*)d_in[8],  (const float*)d_in[12],
                            (const float*)d_in[16], (const float*)d_in[20]};
    const float* atD[4]  = {(const float*)d_in[9],  (const float*)d_in[13],
                            (const float*)d_in[17], (const float*)d_in[21]};
    const float* gatB[4] = {(const float*)d_in[10], (const float*)d_in[14],
                            (const float*)d_in[18], (const float*)d_in[22]};
    const float* lnG[4]  = {(const float*)d_in[23], (const float*)d_in[25],
                            (const float*)d_in[27], (const float*)d_in[29]};
    const float* lnB[4]  = {(const float*)d_in[24], (const float*)d_in[26],
                            (const float*)d_in[28], (const float*)d_in[30]};
    const float* skW = (const float*)d_in[31];
    const float* skB = (const float*)d_in[32];
    const float* fW1 = (const float*)d_in[33]; const float* fB1 = (const float*)d_in[34];
    const float* fW2 = (const float*)d_in[35]; const float* fB2 = (const float*)d_in[36];
    const float* mW1 = (const float*)d_in[37]; const float* mB1 = (const float*)d_in[38];
    const float* mW2 = (const float*)d_in[39]; const float* mB2 = (const float*)d_in[40];
    const float* Wb  = (const float*)d_in[41];
    const float* bb  = (const float*)d_in[42];

    // output regions
    float* f_sc  = (float*)d_out;
    float* m_sc  = f_sc + (size_t)QQ * NNODE;
    float* emb32 = m_sc + (size_t)QQ * NNODE;

    // workspace bump allocator
    char* wp = (char*)d_ws;
    auto alloc = [&](size_t bytes) -> void* {
        void* r = (void*)wp; wp += (bytes + 255) & ~(size_t)255; return r;
    };
    const int Hs[4]  = {8, 8, 8, 1};
    const int Cs[4]  = {32, 32, 32, 128};
    const int HCs[4] = {256, 256, 256, 128};

    _Float16* wip_h = (_Float16*)alloc((size_t)HIDD * FEAT * 2);
    _Float16* gatW_h[4];
    for (int i = 0; i < 4; ++i) gatW_h[i] = (_Float16*)alloc((size_t)HCs[i] * HIDD * 2);
    _Float16* skW_h = (_Float16*)alloc((size_t)EMBD * HIDD * 2);
    _Float16* fW1_h = (_Float16*)alloc((size_t)HIDD * EMBD * 2);
    _Float16* fW2_h = (_Float16*)alloc((size_t)HIDD * HIDD * 2);
    _Float16* mW1_h = (_Float16*)alloc((size_t)HIDD * EMBD * 2);
    _Float16* mW2_h = (_Float16*)alloc((size_t)HIDD * HIDD * 2);
    _Float16* WbT_h = (_Float16*)alloc((size_t)EMBD * HIDD * 2);   // [128,256]
    _Float16* xh    = (_Float16*)alloc((size_t)NNODE * FEAT * 2);
    _Float16* h16   = (_Float16*)alloc((size_t)NNODE * HIDD * 2);
    _Float16* h0_16 = (_Float16*)alloc((size_t)NNODE * HIDD * 2);
    float*    z32   = (float*)alloc((size_t)NNODE * HIDD * 4);
    float*    gout  = (float*)alloc((size_t)NNODE * HIDD * 4);
    float*    a_src = (float*)alloc((size_t)NNODE * 8 * 4);
    float*    a_dst = (float*)alloc((size_t)NNODE * 8 * 4);
    unsigned* m_enc = (unsigned*)alloc((size_t)NNODE * 8 * 4);
    float*    ssum  = (float*)alloc((size_t)NNODE * 8 * 4);
    _Float16* emb16 = (_Float16*)alloc((size_t)NNODE * EMBD * 2);
    _Float16* q16   = (_Float16*)alloc((size_t)QQ * EMBD * 2);
    _Float16* mlp16 = (_Float16*)alloc((size_t)QQ * HIDD * 2);
    _Float16* pq16  = (_Float16*)alloc((size_t)QQ * HIDD * 2);
    _Float16* t16   = (_Float16*)alloc((size_t)QQ * EMBD * 2);

    auto cvt = [&](const float* s, _Float16* dd, long n) {
        k_f2h<<<(unsigned)((n + 255) / 256), 256, 0, stream>>>(s, dd, n);
    };

    // 0) f16 weight / input conversions
    cvt(wip, wip_h, (long)HIDD * FEAT);
    for (int i = 0; i < 4; ++i) cvt(gatW[i], gatW_h[i], (long)HCs[i] * HIDD);
    cvt(skW, skW_h, (long)EMBD * HIDD);
    cvt(fW1, fW1_h, (long)HIDD * EMBD); cvt(fW2, fW2_h, (long)HIDD * HIDD);
    cvt(mW1, mW1_h, (long)HIDD * EMBD); cvt(mW2, mW2_h, (long)HIDD * HIDD);
    k_f2h_t<<<(HIDD * EMBD + 255) / 256, 256, 0, stream>>>(Wb, WbT_h, HIDD, EMBD);
    cvt(x, xh, (long)NNODE * FEAT);

    // 1) h0 = relu(x @ Wip^T + b)   [16384,256]
    k_gemm<1, 0, 0><<<dim3(HIDD / 128, NNODE / 64), 256, 0, stream>>>(
        xh, wip_h, bip, nullptr, h16, NNODE, HIDD, FEAT,
        nullptr, nullptr, nullptr, nullptr, 0);
    hipMemcpyAsync(h0_16, h16, (size_t)NNODE * HIDD * 2, hipMemcpyDeviceToDevice, stream);

    // 2) four GAT layers (+self loops, segment softmax, LayerNorm)
    const int TOT = EEDGE + NNODE;
    for (int i = 0; i < 4; ++i) {
        const int H = Hs[i], C = Cs[i], HC = HCs[i];
        k_gemm<0, 0, 0><<<dim3(HC / 128, NNODE / 64), 256, 0, stream>>>(
            h16, gatW_h[i], nullptr, z32, nullptr, NNODE, HC, HIDD,
            nullptr, nullptr, nullptr, nullptr, 0);
        k_attn<<<(NNODE * H + 255) / 256, 256, 0, stream>>>(
            z32, atS[i], atD[i], a_src, a_dst, NNODE, H, C);
        k_fill0<<<(NNODE * H + 255) / 256, 256, 0, stream>>>(m_enc, (long)NNODE * H);
        k_fill0<<<(NNODE * H + 255) / 256, 256, 0, stream>>>((unsigned*)ssum, (long)NNODE * H);
        long tot = (long)TOT * H;
        k_edge_max<<<(unsigned)((tot + 255) / 256), 256, 0, stream>>>(
            ei, a_src, a_dst, m_enc, EEDGE, TOT, H);
        k_edge_sum<<<(unsigned)((tot + 255) / 256), 256, 0, stream>>>(
            ei, a_src, a_dst, m_enc, ssum, EEDGE, TOT, H);
        k_fill0<<<(unsigned)(((long)NNODE * HC + 255) / 256), 256, 0, stream>>>(
            (unsigned*)gout, (long)NNODE * HC);
        k_edge_acc<<<(unsigned)((tot + 255) / 256), 256, 0, stream>>>(
            ei, a_src, a_dst, m_enc, ssum, z32, gout, EEDGE, TOT, H, C);
        if (i < 3)
            k_ln<<<NNODE / 8, 256, 0, stream>>>(gout, gatB[i], lnG[i], lnB[i],
                                                nullptr, h16, NNODE, HC, 1);
        else
            k_ln<<<NNODE / 8, 256, 0, stream>>>(gout, gatB[i], lnG[i], lnB[i],
                                                emb32, nullptr, NNODE, EMBD, 0);
    }

    // 3) emb = LN_out + h0 @ skipW^T + skip_b  (accumulating GEMM epilogue -> f32+f16)
    k_gemm<0, 1, 0><<<dim3(EMBD / 128, NNODE / 64), 256, 0, stream>>>(
        h0_16, skW_h, skB, emb32, emb16, NNODE, EMBD, HIDD,
        nullptr, nullptr, nullptr, nullptr, 0);

    // 4) q = emb[query_idx]
    k_gather16<<<(QQ * EMBD + 255) / 256, 256, 0, stream>>>(emb16, qidx, q16, QQ, EMBD);

    // 5) father / mother heads: MLP -> @Wb -> @emb^T (masked epilogue)
    const _Float16* W1[2] = {fW1_h, mW1_h};  const float* B1[2] = {fB1, mB1};
    const _Float16* W2[2] = {fW2_h, mW2_h};  const float* B2[2] = {fB2, mB2};
    float* SC[2] = {f_sc, m_sc};
    const int TG[2] = {1, 0};
    for (int p = 0; p < 2; ++p) {
        k_gemm<1, 0, 0><<<dim3(HIDD / 128, QQ / 64), 256, 0, stream>>>(
            q16, W1[p], B1[p], nullptr, mlp16, QQ, HIDD, EMBD,
            nullptr, nullptr, nullptr, nullptr, 0);
        k_gemm<1, 0, 0><<<dim3(HIDD / 128, QQ / 64), 256, 0, stream>>>(
            mlp16, W2[p], B2[p], nullptr, pq16, QQ, HIDD, HIDD,
            nullptr, nullptr, nullptr, nullptr, 0);
        k_gemm<0, 0, 0><<<dim3(EMBD / 128, QQ / 64), 256, 0, stream>>>(
            pq16, WbT_h, nullptr, nullptr, t16, QQ, EMBD, HIDD,
            nullptr, nullptr, nullptr, nullptr, 0);
        // scores = t @ emb^T with fused bilinear bias + gender/generation mask
        k_gemm<0, 0, 1><<<dim3(NNODE / 128, QQ / 64), 256, 0, stream>>>(
            t16, emb16, nullptr, SC[p], nullptr, QQ, NNODE, EMBD,
            gender, gener, qidx, bb, TG[p]);
    }
}
